// KDELoss_53188874993944
// MI455X (gfx1250) — matrix operand order repaired and checked by reading, compiled, not run on hardware
//
#include <hip/hip_runtime.h>
#include <hip/hip_bf16.h>

// ---------------------------------------------------------------------------
// KDE entropy loss, fused fp16-WMMA implementation for gfx1250 (MI455X).
//
//   Xn  = row-l2-normalize(X)            [N, D]  (N=16384, D=256)
//   sim = Xn @ Xn^T                      (never materialized: 1 GB!)
//   density[i] = sum_j exp(5*sim[i][j])
//   out = -mean(log(density + 1e-9))
//
// Compute-bound: 137 GFLOP of GEMM on 16 MB of input -> run the GEMM on
// v_wmma_f32_16x16x32_f16 (fp16 operands are exact enough for normalized
// vectors, fp32 accumulate), fuse exp+rowsum into the tile epilogue, and
// keep everything (8 MB fp16 X + 1 MB partials) in L2. No atomics anywhere
// so results are bitwise deterministic across replays.
// ---------------------------------------------------------------------------

typedef __attribute__((ext_vector_type(16))) _Float16 v16h;
typedef __attribute__((ext_vector_type(8)))  _Float16 v8h;
typedef __attribute__((ext_vector_type(8)))  float    v8f;
typedef __attribute__((ext_vector_type(4)))  float    v4f;

#define D_DIM          256   // feature dim (K of the GEMM)
#define KSTEPS         8     // 256 / 32 (K per wmma_f32_16x16x32_f16)
#define TILE           16
#define NT_PER_WAVE    64    // column tiles handled per wave (-> 16 chunks for N=16384)
#define WAVES_PER_BLK  8     // 256-thread blocks = 8 wave32s

// --------------------------- kernel 1: normalize ---------------------------
// One wave32 per row: 8 fp32 elements per lane, sum-of-squares via lane
// shuffles (wave32-correct masks 16..1), emit fp16 row into workspace.
__global__ void kde_normalize_f16(const float* __restrict__ X,
                                  _Float16* __restrict__ Xh, int N) {
  const int wave = threadIdx.x >> 5;
  const int lane = threadIdx.x & 31;
  const int row  = blockIdx.x * WAVES_PER_BLK + wave;
  if (row >= N) return;

  const float* p = X + (size_t)row * D_DIM + lane * 8;
  v4f a = *(const v4f*)p;
  v4f b = *(const v4f*)(p + 4);

  float s = 0.f;
#pragma unroll
  for (int i = 0; i < 4; ++i) s += a[i] * a[i] + b[i] * b[i];
#pragma unroll
  for (int m = 16; m >= 1; m >>= 1) s += __shfl_xor(s, m, 32);

  const float scale = 1.0f / fmaxf(sqrtf(s), 1e-12f);

  v8h h;
#pragma unroll
  for (int i = 0; i < 4; ++i) {
    h[i]     = (_Float16)(a[i] * scale);
    h[i + 4] = (_Float16)(b[i] * scale);
  }
  *(v8h*)(Xh + (size_t)row * D_DIM + lane * 8) = h;
}

// ----------------- kernel 2: fused WMMA GEMM + exp row-sums ----------------
// Wave w of block (mg,cc) owns row-tile mt = mg*8+w and column-chunk cc
// (64 column tiles). The full K=256 A-strip (16 rows) lives in 64 VGPRs and
// is reused for all 64 tiles; all 8 waves stream the SAME B tiles -> WGP$/L2
// hits. Fragment layouts per CDNA5 ISA 7.12.2 (wave32):
//   A (16x32 f16): lane r=lane&15 is row M=r; hi=lane>>4 selects K-halves:
//       halves 0..7  <- K = k0 + 8*hi .. +7        (one b128 load)
//       halves 8..15 <- K = k0 + 16 + 8*hi .. +7   (one b128 load)
//   B (32x16 f16): lane r is column N=r; halves 0..15 <- same K order as A (two b128 loads)
//       and since B = Xn^T, column N of B == row N of Xn (row-major, no
//       transpose instruction needed).
//   C (16x16 f32): VGPR i, lanes 0-15 -> (M=i, N=lane); lanes 16-31 -> (M=8+i).
__global__ void __launch_bounds__(256)
kde_wmma_tiles(const _Float16* __restrict__ Xh, float* __restrict__ partial,
               int N, int chunks) {
  const int wave = threadIdx.x >> 5;
  const int lane = threadIdx.x & 31;
  const int r    = lane & 15;
  const int hi   = lane >> 4;

  const int cc = blockIdx.x % chunks;           // column chunk
  const int mg = blockIdx.x / chunks;           // row-tile group
  const int mt = mg * WAVES_PER_BLK + wave;     // this wave's 16-row tile

  union Frag { v16h v; v8h h[2]; };

  // Hoist the whole A strip (16 rows x K=256) into registers.
  const _Float16* arow = Xh + (size_t)(mt * TILE + r) * D_DIM;
  Frag A[KSTEPS];
#pragma unroll
  for (int ks = 0; ks < KSTEPS; ++ks) {
    const int k0 = ks * 32;
    A[ks].h[0] = *(const v8h*)(arow + k0 + hi * 8);
    A[ks].h[1] = *(const v8h*)(arow + k0 + 16 + hi * 8);
  }

  float esum[8];
#pragma unroll
  for (int i = 0; i < 8; ++i) esum[i] = 0.f;

  const int nt0 = cc * NT_PER_WAVE;
  for (int t = 0; t < NT_PER_WAVE; ++t) {
    const _Float16* brow =
        Xh + (size_t)((nt0 + t) * TILE + r) * D_DIM + hi * 8;

    v8f c = {};
#pragma unroll
    for (int ks = 0; ks < KSTEPS; ++ks) {
      Frag B;
      B.h[0] = *(const v8h*)(brow + ks * 32);
      B.h[1] = *(const v8h*)(brow + ks * 32 + 16);
      c = __builtin_amdgcn_wmma_f32_16x16x32_f16(
          /*neg_a=*/false, A[ks].v, /*neg_b=*/false, B.v,
          /*c_mod=*/(short)0, c, /*reuse_a=*/false, /*reuse_b=*/false);
    }
    // Fused epilogue: exp(5*sim), accumulated per (row, lane-column) slot.
#pragma unroll
    for (int i = 0; i < 8; ++i) esum[i] += __expf(5.0f * c[i]);
  }

  // Reduce across the 16 lane-columns of each half (xor masks 1..8 stay
  // inside each 16-lane group), then lane r==0 of each half writes the
  // partial row-sum for its chunk. No atomics -> deterministic.
#pragma unroll
  for (int i = 0; i < 8; ++i) {
    float v = esum[i];
    v += __shfl_xor(v, 1, 32);
    v += __shfl_xor(v, 2, 32);
    v += __shfl_xor(v, 4, 32);
    v += __shfl_xor(v, 8, 32);
    if (r == 0) {
      const int row = mt * TILE + hi * 8 + i;
      partial[(size_t)row * chunks + cc] = v;
    }
  }
}

// -------------------- kernel 3: log + mean (single block) ------------------
__global__ void kde_entropy_reduce(const float* __restrict__ partial,
                                   float* __restrict__ out, int N, int chunks) {
  __shared__ float red[256];
  float acc = 0.f;
  for (int row = threadIdx.x; row < N; row += blockDim.x) {
    float d = 0.f;
    for (int c = 0; c < chunks; ++c) d += partial[(size_t)row * chunks + c];
    acc += logf(d + 1e-9f);
  }
  red[threadIdx.x] = acc;
  __syncthreads();
  for (int s = 128; s > 0; s >>= 1) {
    if ((int)threadIdx.x < s) red[threadIdx.x] += red[threadIdx.x + s];
    __syncthreads();
  }
  if (threadIdx.x == 0) out[0] = -red[0] / (float)N;
}

// ------------------------------- launcher ----------------------------------
extern "C" void kernel_launch(void* const* d_in, const int* in_sizes, int n_in,
                              void* d_out, int out_size, void* d_ws,
                              size_t ws_size, hipStream_t stream) {
  const float* X = (const float*)d_in[0];
  float* out = (float*)d_out;

  const int N = in_sizes[0] / D_DIM;                  // 16384
  const int chunks = N / (TILE * NT_PER_WAVE);        // 16

  // Workspace layout: [0, 2*N*D)  fp16 normalized X   (8 MB)
  //                   [8MB, +4*N*chunks) fp32 partial row-sums (1 MB)
  char* ws = (char*)d_ws;
  _Float16* Xh   = (_Float16*)ws;
  float* partial = (float*)(ws + (size_t)N * D_DIM * sizeof(_Float16));

  // 1) normalize -> fp16
  kde_normalize_f16<<<(N + WAVES_PER_BLK - 1) / WAVES_PER_BLK, 256, 0,
                      stream>>>(X, Xh, N);

  // 2) fused WMMA GEMM + exp + partial row-sums
  const int row_groups = N / (TILE * WAVES_PER_BLK);  // 128
  kde_wmma_tiles<<<row_groups * chunks, 256, 0, stream>>>(Xh, partial, N,
                                                          chunks);

  // 3) log + mean -> scalar entropy
  kde_entropy_reduce<<<1, 256, 0, stream>>>(partial, out, N, chunks);
}
